// AutoEncoder_30648886624860
// MI455X (gfx1250) — compile-verified, matrix operand order checked by reading, not run on hardware
//
#include <hip/hip_runtime.h>
#include <math.h>

// ---------------------------------------------------------------------------
// GRU autoencoder for MI455X (gfx1250, wave32).
// - All GEMM operands pre-converted to bf16 once (weights at launch start,
//   activations produced in bf16 by the gather/gate kernels).
// - One bf16-WMMA GEMM kernel (v_wmma_f32_16x16x32_bf16), 128x64 block tile,
//   8 waves x (32x32), double-buffered LDS staging via async global->LDS
//   copies (ASYNCcnt-tracked global_load_async_to_lds_b128).
// - Dominant cost: logits GEMM [14336,512]x[32000,512]^T, ~470 GFLOP and
//   1.83 GB of stores -> store-bandwidth bound at 23.3 TB/s.
// ---------------------------------------------------------------------------

typedef __bf16 bf16;
typedef __attribute__((ext_vector_type(16))) __bf16 v16bf;
typedef __attribute__((ext_vector_type(8)))  float  v8f;
typedef int v4i __attribute__((vector_size(16)));   // matches builtin proto

#define BM 128
#define BN 64
#define BK 32
#define LDSS 40   // padded LDS row stride in bf16 elems (80B) - conflict-free

#define AS1 __attribute__((address_space(1)))
#define AS3 __attribute__((address_space(3)))

#if defined(__has_builtin)
#if __has_builtin(__builtin_amdgcn_global_load_async_to_lds_b128)
#define HAS_ASYNC_LDS 1
#endif
#endif
#ifndef HAS_ASYNC_LDS
#define HAS_ASYNC_LDS 0
#endif

#if HAS_ASYNC_LDS
#if defined(__has_builtin) && __has_builtin(__builtin_amdgcn_s_wait_asynccnt)
#define ASYNC_WAIT(n) __builtin_amdgcn_s_wait_asynccnt(n)
#else
#define ASYNC_WAIT(n) asm volatile("s_wait_asynccnt " #n ::: "memory")
#endif
#else
#define ASYNC_WAIT(n)  // fallback: __syncthreads' dscnt wait orders LDS stores
#endif

__device__ __forceinline__ bf16 f2bf(float f) {
  unsigned u = __builtin_bit_cast(unsigned, f);
  unsigned r = u + 0x7FFFu + ((u >> 16) & 1u);   // round-to-nearest-even
  unsigned short h = (unsigned short)(r >> 16);
  return __builtin_bit_cast(bf16, h);
}

// 16B global->LDS copy: async (ASYNCcnt-tracked) if available, else sync.
__device__ __forceinline__ void cp16(const bf16* __restrict__ g,
                                     bf16* __restrict__ l) {
#if HAS_ASYNC_LDS
  __builtin_amdgcn_global_load_async_to_lds_b128(
      (AS1 v4i*)g, (AS3 v4i*)l, 0, 0);
#else
  *(uint4*)l = *(const uint4*)g;
#endif
}

// 16-bit A (16x32) / B (32x16) fragment per ISA layout:
// lane&15 selects the M (resp. N) row; lanes 0-15 hold K 0-7 & 16-23,
// lanes 16-31 hold K 8-15 & 24-31.
__device__ __forceinline__ v16bf frag_load(const bf16* tile, int row0, int lane) {
  const bf16* p = tile + (row0 + (lane & 15)) * LDSS;
  const int hi = (lane >> 4) << 3;
  v16bf f;
#pragma unroll
  for (int e = 0; e < 8; ++e) {
    f[e]     = p[hi + e];
    f[8 + e] = p[16 + hi + e];
  }
  return f;
}

// C[M,N] = act(A[M,K] * W[N,K]^T + bias), A/W bf16 row-major, f32 accumulate.
// Optional bf16 shadow store Cb (same layout as C).
// Requires M%128==0, N%64==0, K%32==0, lda/ldw multiples of 8 (all hold).
// swapTB: row m = t*Bdim+b stored at C[b*Tdim*ldc + t*ldc + n] (for re_emb).
__global__ __launch_bounds__(256)
void gemm_nt_wmma(const bf16* __restrict__ A, int lda,
                  const bf16* __restrict__ W, int ldw,
                  const float* __restrict__ bias,
                  float* __restrict__ C, bf16* __restrict__ Cb, int ldc,
                  int K, int act, int swapTB, int Tdim, int Bdim) {
  __shared__ __align__(16) bf16 As[2][BM * LDSS];
  __shared__ __align__(16) bf16 Bs[2][BN * LDSS];

  const int tid  = threadIdx.x;
  const int lane = tid & 31;
  const int wave = tid >> 5;          // 8 waves: 4 along M x 2 along N
  const int wm   = (wave & 3) * 32;
  const int wn   = (wave >> 2) * 32;
  const size_t rowA0 = (size_t)blockIdx.y * BM;
  const size_t rowB0 = (size_t)blockIdx.x * BN;
  const bf16* Ab = A + rowA0 * (size_t)lda;
  const bf16* Wb = W + rowB0 * (size_t)ldw;

  // Per tile: A = 128 rows x 4 16B-chunks (2 per thread), B = 64 x 4 (1/thread)
  // -> 3 async instructions per wave per tile.
  auto stage = [&](int buf, int k0) {
#pragma unroll
    for (int it2 = 0; it2 < 2; ++it2) {
      int idx = tid + (it2 << 8);
      int r = idx >> 2, c = (idx & 3) << 3;
      cp16(Ab + (size_t)r * lda + k0 + c, &As[buf][r * LDSS + c]);
    }
    {
      int r = tid >> 2, c = (tid & 3) << 3;
      cp16(Wb + (size_t)r * ldw + k0 + c, &Bs[buf][r * LDSS + c]);
    }
  };

  v8f acc[2][2] = {};
  const int nt = K >> 5;

  stage(0, 0);
  for (int it = 0; it < nt; ++it) {
    const int cur = it & 1;
    if (it + 1 < nt) {
#if !HAS_ASYNC_LDS
      __builtin_prefetch(Ab + (size_t)(tid & 127) * lda + ((it + 1) << 5), 0, 1);
#endif
      stage(cur ^ 1, (it + 1) << 5);   // fill other buffer while computing
      ASYNC_WAIT(3);                   // 3 newly-issued ops may stay in flight
    } else {
      ASYNC_WAIT(0);
    }
    __syncthreads();                   // staged tile visible to all waves

    const bf16* as = As[cur];
    const bf16* bs = Bs[cur];
    v16bf a0 = frag_load(as, wm,      lane);
    v16bf a1 = frag_load(as, wm + 16, lane);
    v16bf b0 = frag_load(bs, wn,      lane);
    v16bf b1 = frag_load(bs, wn + 16, lane);
    acc[0][0] = __builtin_amdgcn_wmma_f32_16x16x32_bf16(false, a0, false, b0,
                  (short)0, acc[0][0], false, false);
    acc[0][1] = __builtin_amdgcn_wmma_f32_16x16x32_bf16(false, a0, false, b1,
                  (short)0, acc[0][1], false, false);
    acc[1][0] = __builtin_amdgcn_wmma_f32_16x16x32_bf16(false, a1, false, b0,
                  (short)0, acc[1][0], false, false);
    acc[1][1] = __builtin_amdgcn_wmma_f32_16x16x32_bf16(false, a1, false, b1,
                  (short)0, acc[1][1], false, false);
    __syncthreads();                   // protect buffer refilled next iter
  }

  // f32 C/D layout: lane&15 = N, (lane>>4)*8 + vgpr = M.
#pragma unroll
  for (int i = 0; i < 2; ++i) {
#pragma unroll
    for (int j = 0; j < 2; ++j) {
      const int mbase = (int)rowA0 + wm + i * 16 + ((lane >> 4) << 3);
      const int nn    = (int)rowB0 + wn + j * 16 + (lane & 15);
      const float bv  = bias ? bias[nn] : 0.0f;
#pragma unroll
      for (int r = 0; r < 8; ++r) {
        int m = mbase + r;
        float v = acc[i][j][r] + bv;
        if (act) v = (v >= 0.0f) ? v : 0.02f * v;   // LeakyReLU(0.02)
        size_t off;
        if (swapTB) {
          int t = m / Bdim, b = m - t * Bdim;
          off = ((size_t)b * Tdim + t) * (size_t)ldc + nn;
        } else {
          off = (size_t)m * (size_t)ldc + nn;
        }
        C[off] = v;
        if (Cb) Cb[off] = f2bf(v);
      }
    }
  }
}

// One-shot f32 -> bf16 conversion (n multiple of 4).
__global__ void cvt_bf16(const float* __restrict__ in, bf16* __restrict__ out,
                         int n) {
  int i = (blockIdx.x * blockDim.x + threadIdx.x) << 2;
  if (i >= n) return;
  float4 v = *(const float4*)(in + i);
  out[i] = f2bf(v.x); out[i + 1] = f2bf(v.y);
  out[i + 2] = f2bf(v.z); out[i + 3] = f2bf(v.w);
}

// Fused GRU gate math (torch order r,z,n). gh==nullptr -> hidden gates = bhh
// (h=0 fast path). hprev==nullptr -> 0. Writes f32 h (next hprev) and
// optionally a bf16 shadow (next GEMM A operand) and a bf16 ctx scatter
// (always when x_lens==nullptr, else only when t == x_lens[b]-1).
__global__ void gru_gate(const float* __restrict__ gi, const float* __restrict__ gh,
                         const float* __restrict__ bhh, const float* __restrict__ hprev,
                         float* __restrict__ hout, bf16* __restrict__ hout_bf,
                         bf16* __restrict__ ctx, int ctx_stride, int ctx_off,
                         const int* __restrict__ x_lens, int t, int B, int H) {
  int idx = blockIdx.x * blockDim.x + threadIdx.x;
  if (idx >= B * H) return;
  int b = idx / H, j = idx - b * H;
  const float* gib = gi + (size_t)b * 3 * H;
  float ir = gib[j], iz = gib[H + j], in = gib[2 * H + j];
  float hr, hz, hn;
  if (gh) {
    const float* ghb = gh + (size_t)b * 3 * H;
    hr = ghb[j]; hz = ghb[H + j]; hn = ghb[2 * H + j];
  } else {
    hr = bhh[j]; hz = bhh[H + j]; hn = bhh[2 * H + j];
  }
  float r = 1.0f / (1.0f + __expf(-(ir + hr)));
  float z = 1.0f / (1.0f + __expf(-(iz + hz)));
  float n = tanhf(in + r * hn);
  float hp = hprev ? hprev[(size_t)b * H + j] : 0.0f;
  float h = (1.0f - z) * n + z * hp;
  hout[(size_t)b * H + j] = h;
  if (hout_bf) hout_bf[(size_t)b * H + j] = f2bf(h);
  if (ctx) {
    if (!x_lens || (x_lens[b] - 1 == t))
      ctx[(size_t)b * ctx_stride + ctx_off + j] = f2bf(h);
  }
}

// embedded[b,t,:] = bf16(emb[x[b,t],:])
__global__ void embed_gather(const int* __restrict__ x, const float* __restrict__ emb,
                             bf16* __restrict__ out, int E) {
  int bt = blockIdx.x;                // B*T blocks
  int e  = threadIdx.x;               // E threads
  int tok = x[bt];
  out[(size_t)bt * E + e] = f2bf(emb[(size_t)tok * E + e]);
}

__global__ void last_gather(const bf16* __restrict__ embedded,
                            const int* __restrict__ x_lens,
                            bf16* __restrict__ xlast, int T, int E) {
  int b = blockIdx.x, e = threadIdx.x;
  int idx = x_lens[b] - 1;
  xlast[(size_t)b * E + e] = embedded[((size_t)b * T + idx) * E + e];
}

extern "C" void kernel_launch(void* const* d_in, const int* in_sizes, int n_in,
                              void* d_out, int out_size, void* d_ws, size_t ws_size,
                              hipStream_t stream) {
  const int B = 512, T = 28, V = 32000, E = 256, H = 512, D = 512, L = 128;
  (void)in_sizes; (void)n_in; (void)out_size; (void)ws_size;

  const int*   x         = (const int*)d_in[0];
  const int*   x_lens    = (const int*)d_in[1];
  const float* emb       = (const float*)d_in[2];
  const float* enc_Wih_f = (const float*)d_in[3];
  const float* enc_Whh_f = (const float*)d_in[4];
  const float* enc_bih_f = (const float*)d_in[5];
  const float* enc_bhh_f = (const float*)d_in[6];
  const float* enc_Wih_b = (const float*)d_in[7];
  const float* enc_bih_b = (const float*)d_in[9];
  const float* enc_bhh_b = (const float*)d_in[10];
  const float* fc1_w     = (const float*)d_in[11];
  const float* fc1_b     = (const float*)d_in[12];
  const float* fc2_w     = (const float*)d_in[13];
  const float* fc2_b     = (const float*)d_in[14];
  const float* dec_Wih   = (const float*)d_in[15];
  const float* dec_Whh   = (const float*)d_in[16];
  const float* dec_bih   = (const float*)d_in[17];
  const float* dec_bhh   = (const float*)d_in[18];
  const float* re_w      = (const float*)d_in[19];
  const float* re_b      = (const float*)d_in[20];
  const float* hv_w      = (const float*)d_in[21];
  const float* hv_b      = (const float*)d_in[22];

  float* out_logits = (float*)d_out;                       // [T,B,V]
  float* out_z      = out_logits + (size_t)T * B * V;      // [B,L]
  float* out_re     = out_z + (size_t)B * L;               // [B,T,E]

  // ----- workspace: f32 region (~42 MB), then bf16 region (~64 MB) -----
  float* ws = (float*)d_ws;
  float* gi    = ws; ws += (size_t)B * 3 * H;
  float* gh    = ws; ws += (size_t)B * 3 * H;
  float* gidec = ws; ws += (size_t)B * 3 * D;
  float* hbuf  = ws; ws += (size_t)B * H;
  float* out0  = ws; ws += (size_t)B * D;
  float* outs  = ws; ws += (size_t)T * B * D;   // [T,B,D]
  float* hb    = ws; ws += (size_t)B * H;

  bf16* wb = (bf16*)ws;
  bf16* bWih_f = wb; wb += (size_t)3 * H * E;
  bf16* bWhh_f = wb; wb += (size_t)3 * H * H;
  bf16* bWih_b = wb; wb += (size_t)3 * H * E;
  bf16* bFc1   = wb; wb += (size_t)L * 2 * H;
  bf16* bFc2   = wb; wb += (size_t)D * L;
  bf16* bDWih  = wb; wb += (size_t)3 * D * D;
  bf16* bDWhh  = wb; wb += (size_t)3 * D * D;
  bf16* bRe    = wb; wb += (size_t)E * D;
  bf16* bHv    = wb; wb += (size_t)V * D;
  bf16* bEmbd  = wb; wb += (size_t)B * T * E;   // embedded, bf16
  bf16* bXlast = wb; wb += (size_t)B * E;
  bf16* bH     = wb; wb += (size_t)B * H;
  bf16* bCtx   = wb; wb += (size_t)B * 2 * H;
  bf16* bZ     = wb; wb += (size_t)B * L;
  bf16* bOut0  = wb; wb += (size_t)B * D;
  bf16* bOuts  = wb; wb += (size_t)T * B * D;

  auto cvt = [&](const float* s, bf16* d, size_t n) {
    cvt_bf16<<<dim3((unsigned)((n / 4 + 255) / 256)), 256, 0, stream>>>(s, d, (int)n);
  };
  auto gemm = [&](const bf16* A, int lda, const bf16* W, int ldw,
                  const float* bias, float* C, bf16* Cb, int ldc,
                  int M, int N, int K, int act = 0, int swapTB = 0,
                  int Td = 0, int Bd = 0) {
    gemm_nt_wmma<<<dim3(N / BN, M / BM), 256, 0, stream>>>(
        A, lda, W, ldw, bias, C, Cb, ldc, K, act, swapTB, Td, Bd);
  };
  const int gateBlocks = (B * H + 255) / 256;

  // 0) One-shot weight conversion to bf16.
  cvt(enc_Wih_f, bWih_f, (size_t)3 * H * E);
  cvt(enc_Whh_f, bWhh_f, (size_t)3 * H * H);
  cvt(enc_Wih_b, bWih_b, (size_t)3 * H * E);
  cvt(fc1_w, bFc1, (size_t)L * 2 * H);
  cvt(fc2_w, bFc2, (size_t)D * L);
  cvt(dec_Wih, bDWih, (size_t)3 * D * D);
  cvt(dec_Whh, bDWhh, (size_t)3 * D * D);
  cvt(re_w, bRe, (size_t)E * D);
  cvt(hv_w, bHv, (size_t)V * D);

  // 1) Embedding lookup (bf16 out) + last-valid-token gather.
  embed_gather<<<dim3(B * T), E, 0, stream>>>(x, emb, bEmbd, E);
  last_gather<<<dim3(B), E, 0, stream>>>(bEmbd, x_lens, bXlast, T, E);

  // 2) Forward encoder GRU: 28 sequential steps; ctx_f scattered at len-1.
  for (int t = 0; t < T; ++t) {
    gemm(bEmbd + (size_t)t * E, T * E, bWih_f, E, enc_bih_f,
         gi, nullptr, 3 * H, B, 3 * H, E);
    if (t > 0)
      gemm(bH, H, bWhh_f, H, enc_bhh_f, gh, nullptr, 3 * H, B, 3 * H, H);
    gru_gate<<<gateBlocks, 256, 0, stream>>>(
        gi, (t > 0) ? gh : nullptr, enc_bhh_f, (t > 0) ? hbuf : nullptr,
        hbuf, bH, bCtx, 2 * H, 0, x_lens, t, B, H);
  }

  // 3) Backward direction: single GRU step on last valid token, h0=0.
  gemm(bXlast, E, bWih_b, E, enc_bih_b, gi, nullptr, 3 * H, B, 3 * H, E);
  gru_gate<<<gateBlocks, 256, 0, stream>>>(
      gi, nullptr, enc_bhh_b, nullptr, hb, nullptr, bCtx, 2 * H, H,
      nullptr, -1, B, H);

  // 4) fc1 -> z (f32 output + bf16 shadow), fc2 -> out0 (f32 + bf16 shadow).
  gemm(bCtx, 2 * H, bFc1, 2 * H, fc1_b, out_z, bZ, L, B, L, 2 * H);
  gemm(bZ, L, bFc2, L, fc2_b, out0, bOut0, D, B, D, L);

  // 5) Decoder: input gates constant across steps -> precompute once.
  gemm(bOut0, D, bDWih, D, dec_bih, gidec, nullptr, 3 * D, B, 3 * D, D);
  for (int t = 0; t < T; ++t) {
    const bf16*  hprev_bf = (t == 0) ? bOut0 : (bOuts + (size_t)(t - 1) * B * D);
    const float* hprev    = (t == 0) ? out0  : (outs + (size_t)(t - 1) * B * D);
    gemm(hprev_bf, D, bDWhh, D, dec_bhh, gh, nullptr, 3 * D, B, 3 * D, D);
    gru_gate<<<gateBlocks, 256, 0, stream>>>(
        gidec, gh, nullptr, hprev, outs + (size_t)t * B * D,
        bOuts + (size_t)t * B * D, nullptr, 0, 0, nullptr, -1, B, D);
  }

  // 6) re_emb = leakyrelu(outs @ re_w^T + re_b), stored [B,T,E].
  gemm(bOuts, D, bRe, D, re_b, out_re, nullptr, E, T * B, E, D,
       /*act=*/1, /*swapTB=*/1, T, B);

  // 7) logits = outs @ hv_w^T + hv_b, stored [T,B,V]. The big one.
  gemm(bOuts, D, bHv, D, hv_b, out_logits, nullptr, V, T * B, V, D);
}
